// QuantumClassifier_59725815218737
// MI455X (gfx1250) — matrix operand order repaired and checked by reading, not compiled, not run on hardware
//
#include <hip/hip_runtime.h>
#include <hip/hip_bf16.h>

typedef __attribute__((ext_vector_type(16))) _Float16 v16h;
typedef __attribute__((ext_vector_type(8)))  float    v8f;

namespace {

constexpr int KDIM = 256;   // input features
constexpr int HDIM = 64;    // hidden features
constexpr int ADIM = 4;     // angles (= N_QUBITS)
constexpr int WAVES = 8;    // waves per block (256 threads, wave32)
constexpr int ROWS_PER_WAVE = 16;
constexpr int ROWS_PER_BLOCK = WAVES * ROWS_PER_WAVE;  // 128
constexpr int HPAD = HDIM + 1;                          // 65 -> conflict-free LDS rows

// ---------- 4-qubit statevector gates (16 complex amps in registers) ----------
// wire i <-> flat-index bit (8 >> i); all indices compile-time via template mask.

template<int M>
__device__ __forceinline__ void qry(float (&sr)[16], float (&si)[16], float th) {
  float s, c;
  __sincosf(0.5f * th, &s, &c);
  #pragma unroll
  for (int i = 0; i < 16; ++i)
    if ((i & M) == 0) {
      const int j = i | M;
      const float r0 = sr[i], r1 = sr[j], m0 = si[i], m1 = si[j];
      sr[i] = c * r0 - s * r1;
      sr[j] = s * r0 + c * r1;
      si[i] = c * m0 - s * m1;
      si[j] = s * m0 + c * m1;
    }
}

template<int M>
__device__ __forceinline__ void qrz(float (&sr)[16], float (&si)[16], float th) {
  // diag(e^{-i th/2}, e^{+i th/2})
  float s, c;
  __sincosf(0.5f * th, &s, &c);
  #pragma unroll
  for (int i = 0; i < 16; ++i) {
    const float sg = (i & M) ? s : -s;
    const float r = sr[i], m = si[i];
    sr[i] = r * c - m * sg;
    si[i] = m * c + r * sg;
  }
}

template<int C, int T>
__device__ __forceinline__ void qcnot(float (&sr)[16], float (&si)[16]) {
  #pragma unroll
  for (int i = 0; i < 16; ++i)
    if ((i & C) && !(i & T)) {
      const int j = i | T;
      float t;
      t = sr[i]; sr[i] = sr[j]; sr[j] = t;
      t = si[i]; si[i] = si[j]; si[j] = t;
    }
}

} // namespace

__global__ __launch_bounds__(256, 1)
void qc_fused_kernel(const float* __restrict__ x,   // [B,256]
                     const float* __restrict__ W1,  // [64,256]
                     const float* __restrict__ b1,  // [64]
                     const float* __restrict__ W2,  // [4,64]
                     const float* __restrict__ b2,  // [4]
                     const float* __restrict__ qw,  // [2,4,2]
                     float* __restrict__ out,       // [B]
                     int B)
{
  __shared__ __align__(32) _Float16 sW1[HDIM * KDIM];        // 32 KB, f16 W1 (row-major [64][256])
  __shared__ float sW2[ADIM * HDIM];                         // 1 KB
  __shared__ float sB1[HDIM];
  __shared__ float sB2[ADIM];
  __shared__ float sH[WAVES * ROWS_PER_WAVE * HPAD];         // 33.3 KB h scratch

  const int tid = threadIdx.x;

  // ---- stage weights into LDS (once per block; W1 lives in L2 across blocks) ----
  for (int i = tid; i < HDIM * KDIM; i += 256) sW1[i] = (_Float16)W1[i];
  for (int i = tid; i < ADIM * HDIM; i += 256) sW2[i] = W2[i];
  if (tid < HDIM) sB1[tid] = b1[tid];
  if (tid < ADIM) sB2[tid] = b2[tid];
  __syncthreads();

  const int wave = tid >> 5;
  const int lane = tid & 31;
  const int hl   = lane & 15;   // row within M-tile / col within N-tile
  const int g    = lane >> 4;   // lane-group per WMMA fragment layout

  const int rowBase = blockIdx.x * ROWS_PER_BLOCK + wave * ROWS_PER_WAVE;
  int arow = rowBase + hl;
  if (arow >= B) arow = B - 1;   // clamp (B is a multiple of 128 in practice)
  const float* xrow = x + (size_t)arow * KDIM;

  // ---- GEMM1: h[16,64] = relu(x_tile @ W1^T + b1) via v_wmma_f32_16x16x32_f16 ----
  v8f acc[4] = { v8f{}, v8f{}, v8f{}, v8f{} };

  #pragma unroll
  for (int k0 = 0; k0 < KDIM; k0 += 32) {
    // A fragment (16x32 f16): lane group g needs K = k0+g*16..+15 (matches B)
    const float4* pa = reinterpret_cast<const float4*>(xrow + k0 + g * 16);
    const float4* pb = reinterpret_cast<const float4*>(xrow + k0 + g * 16 + 8);
    const float4 f0 = pa[0], f1 = pa[1];
    const float4 f2 = pb[0], f3 = pb[1];
    v16h a;
    a[0]  = (_Float16)f0.x; a[1]  = (_Float16)f0.y; a[2]  = (_Float16)f0.z; a[3]  = (_Float16)f0.w;
    a[4]  = (_Float16)f1.x; a[5]  = (_Float16)f1.y; a[6]  = (_Float16)f1.z; a[7]  = (_Float16)f1.w;
    a[8]  = (_Float16)f2.x; a[9]  = (_Float16)f2.y; a[10] = (_Float16)f2.z; a[11] = (_Float16)f2.w;
    a[12] = (_Float16)f3.x; a[13] = (_Float16)f3.y; a[14] = (_Float16)f3.z; a[15] = (_Float16)f3.w;

    #pragma unroll
    for (int t = 0; t < 4; ++t) {
      // B fragment (32x16 f16): lane = output feature t*16+hl, K = k0+g*16..+15 (32B aligned)
      const v16h bfrag = *reinterpret_cast<const v16h*>(
          sW1 + (t * 16 + hl) * KDIM + k0 + g * 16);
      acc[t] = __builtin_amdgcn_wmma_f32_16x16x32_f16(
          false, a, false, bfrag, (short)0, acc[t], false, false);
    }
  }

  // ---- bias + relu, stage h tile into padded LDS (C layout: reg r -> row r+8g) ----
  float* hbase = sH + (size_t)wave * ROWS_PER_WAVE * HPAD;
  #pragma unroll
  for (int t = 0; t < 4; ++t) {
    const float bias = sB1[t * 16 + hl];
    #pragma unroll
    for (int r = 0; r < 8; ++r) {
      float v = acc[t][r] + bias;
      v = fmaxf(v, 0.0f);
      hbase[(r + 8 * g) * HPAD + t * 16 + hl] = v;
    }
  }
  asm volatile("s_wait_dscnt 0" ::: "memory");  // same-wave LDS RAW fence

  // ---- GEMM2 + tanh: each half-wave covers half of K=64, reduce across halves ----
  const float* hrow = hbase + hl * HPAD + g * 32;
  float p0 = 0.f, p1 = 0.f, p2 = 0.f, p3 = 0.f;
  #pragma unroll
  for (int k = 0; k < 32; ++k) {
    const int kk = g * 32 + k;
    const float hv = hrow[k];
    p0 = fmaf(hv, sW2[0 * HDIM + kk], p0);
    p1 = fmaf(hv, sW2[1 * HDIM + kk], p1);
    p2 = fmaf(hv, sW2[2 * HDIM + kk], p2);
    p3 = fmaf(hv, sW2[3 * HDIM + kk], p3);
  }
  p0 += __shfl_xor(p0, 16, 32);
  p1 += __shfl_xor(p1, 16, 32);
  p2 += __shfl_xor(p2, 16, 32);
  p3 += __shfl_xor(p3, 16, 32);

  // ---- 4-qubit circuit: one sample per lane (lanes 0..15) ----
  if (g == 0) {
    const float a0 = tanhf(p0 + sB2[0]);
    const float a1 = tanhf(p1 + sB2[1]);
    const float a2 = tanhf(p2 + sB2[2]);
    const float a3 = tanhf(p3 + sB2[3]);

    float qwl[16];
    #pragma unroll
    for (int i = 0; i < 16; ++i) qwl[i] = qw[i];  // (2,4,2) flat

    float sr[16], si[16];
    #pragma unroll
    for (int i = 0; i < 16; ++i) { sr[i] = 0.f; si[i] = 0.f; }
    sr[0] = 1.f;

    // encoding layer
    qry<8>(sr, si, a0); qry<4>(sr, si, a1);
    qry<2>(sr, si, a2); qry<1>(sr, si, a3);

    // variational layers
    #pragma unroll
    for (int l = 0; l < 2; ++l) {
      const int o = l * 8;
      qry<8>(sr, si, qwl[o + 0]); qrz<8>(sr, si, qwl[o + 1]);
      qry<4>(sr, si, qwl[o + 2]); qrz<4>(sr, si, qwl[o + 3]);
      qry<2>(sr, si, qwl[o + 4]); qrz<2>(sr, si, qwl[o + 5]);
      qry<1>(sr, si, qwl[o + 6]); qrz<1>(sr, si, qwl[o + 7]);
      qcnot<8, 4>(sr, si);
      qcnot<4, 2>(sr, si);
      qcnot<2, 1>(sr, si);
      qcnot<1, 8>(sr, si);
    }

    // <Z_0>: wire 0 is bit 8 of the flat index
    float e = 0.f;
    #pragma unroll
    for (int i = 0; i < 16; ++i) {
      const float p = sr[i] * sr[i] + si[i] * si[i];
      e += (i < 8) ? p : -p;
    }

    const int orow = rowBase + hl;
    if (orow < B) out[orow] = e;
  }
}

extern "C" void kernel_launch(void* const* d_in, const int* in_sizes, int n_in,
                              void* d_out, int out_size, void* d_ws, size_t ws_size,
                              hipStream_t stream) {
  (void)n_in; (void)d_ws; (void)ws_size; (void)out_size;
  const float* x  = (const float*)d_in[0];
  const float* W1 = (const float*)d_in[1];
  const float* b1 = (const float*)d_in[2];
  const float* W2 = (const float*)d_in[3];
  const float* b2 = (const float*)d_in[4];
  const float* qw = (const float*)d_in[5];
  float* out = (float*)d_out;

  const int B = in_sizes[0] / KDIM;                 // 65536
  const int blocks = (B + ROWS_PER_BLOCK - 1) / ROWS_PER_BLOCK;  // 512
  qc_fused_kernel<<<blocks, 256, 0, stream>>>(x, W1, b1, W2, b2, qw, out, B);
}
